// MIP_58815282151900
// MI455X (gfx1250) — matrix operand order repaired and checked by reading, not compiled, not run on hardware
//
#include <hip/hip_runtime.h>
#include <hip/hip_bf16.h>

typedef __attribute__((ext_vector_type(8)))  _Float16 h8;
typedef __attribute__((ext_vector_type(16))) _Float16 v16h;
typedef __attribute__((ext_vector_type(8)))  float    v8f;

#define DEV __device__ __forceinline__

// B=8, T=12, N=1024, C0=32, M=64, L=3, C=64, FF=256, heads=4, hd=16, topK=4
// S = B*T = 96 slices; ROWS = S*N = 98304 activation rows.

DEV float wred_max(float v){ for(int o=16;o>0;o>>=1) v=fmaxf(v,__shfl_xor(v,o)); return v; }
DEV float wred_sum(float v){ for(int o=16;o>0;o>>=1) v+=__shfl_xor(v,o); return v; }

// ---------------------------------------------------------------------------
// K0: input_proj + memory-gate top-4 retrieval.  One wave per row.
// Writes H f32 [row][64], HT f16 ch-major [s][64][n], CCAT col 0..63 f16.
// ---------------------------------------------------------------------------
__global__ void k_prep(const float* __restrict__ hist,
                       const float* __restrict__ W_in, const float* __restrict__ b_in,
                       const float* __restrict__ mem1,
                       const float* __restrict__ Wqm, const float* __restrict__ bqm,
                       float* __restrict__ H, _Float16* __restrict__ HT,
                       _Float16* __restrict__ CCAT)
{
    __shared__ float sx[8][32];
    __shared__ float sq[8][32];
    __shared__ float sc[8][64];
    const int w = threadIdx.x >> 5, lane = threadIdx.x & 31;
    const int row = blockIdx.x * 8 + w;
    const int s = row >> 10, n = row & 1023;

    const float h0 = hist[row*3+0], h1 = hist[row*3+1], h2 = hist[row*3+2];
    const float x = h0*W_in[lane] + h1*W_in[32+lane] + h2*W_in[64+lane] + b_in[lane];
    sx[w][lane] = x;
    __syncthreads();

    float q = bqm[lane];
    #pragma unroll 8
    for (int c = 0; c < 32; ++c) q += sx[w][c] * Wqm[c*32 + lane];
    sq[w][lane] = q;
    __syncthreads();

    float l0 = 0.f, l1 = 0.f;
    #pragma unroll 8
    for (int c = 0; c < 32; ++c) {
        const float qc = sq[w][c];
        l0 += qc * mem1[lane*32 + c];
        l1 += qc * mem1[(lane+32)*32 + c];
    }
    const float mx = wred_max(fmaxf(l0, l1));
    sc[w][lane]      = __expf(l0 - mx);   // top-K weights renormalize -> denom cancels
    sc[w][lane + 32] = __expf(l1 - mx);
    __syncthreads();

    float vk[4]; int ik[4]; float wsum = 0.f;
    for (int k = 0; k < 4; ++k) {
        const float a = sc[w][lane], b2 = sc[w][lane + 32];
        float bv; int bi;
        if (b2 > a) { bv = b2; bi = lane + 32; } else { bv = a; bi = lane; }
        for (int o = 16; o > 0; o >>= 1) {
            const float ov = __shfl_xor(bv, o);
            const int   oi = __shfl_xor(bi, o);
            if (ov > bv || (ov == bv && oi < bi)) { bv = ov; bi = oi; }
        }
        vk[k] = bv; ik[k] = bi; wsum += bv;
        if (lane == 0) sc[w][bi] = -1.0f;
        __syncthreads();
    }
    const float inv = 1.0f / wsum;
    float v1 = 0.f;
    #pragma unroll
    for (int k = 0; k < 4; ++k) v1 += vk[k] * inv * mem1[ik[k]*32 + lane];

    H[(size_t)row*64 + lane]      = x;
    H[(size_t)row*64 + 32 + lane] = v1;
    HT[((size_t)s*64 + lane)*1024 + n]      = (_Float16)x;
    HT[((size_t)s*64 + 32 + lane)*1024 + n] = (_Float16)v1;
    CCAT[(size_t)row*256 + lane]      = (_Float16)x;
    CCAT[(size_t)row*256 + 32 + lane] = (_Float16)v1;
}

// ---------------------------------------------------------------------------
// K1: adjacency A = softmax(relu(E1 @ E2^T)) -> f16 row-major [1024][1024].
// ---------------------------------------------------------------------------
__global__ void k_adj(const float* __restrict__ E1, const float* __restrict__ E2,
                      _Float16* __restrict__ A16)
{
    __shared__ float e1r[16];
    __shared__ float red[256];
    const int n = blockIdx.x, tid = threadIdx.x;
    if (tid < 16) e1r[tid] = E1[n*16 + tid];
    __syncthreads();

    float v[4];
    #pragma unroll
    for (int j = 0; j < 4; ++j) {
        const int m = tid + j*256;
        float a = 0.f;
        #pragma unroll
        for (int d = 0; d < 16; ++d) a += e1r[d] * E2[m*16 + d];
        v[j] = fmaxf(a, 0.0f);
    }
    float lm = fmaxf(fmaxf(v[0], v[1]), fmaxf(v[2], v[3]));
    red[tid] = lm; __syncthreads();
    for (int s2 = 128; s2 > 0; s2 >>= 1) {
        if (tid < s2) red[tid] = fmaxf(red[tid], red[tid+s2]);
        __syncthreads();
    }
    const float mx = red[0];
    __syncthreads();
    float e[4]; float ls = 0.f;
    #pragma unroll
    for (int j = 0; j < 4; ++j) { e[j] = __expf(v[j] - mx); ls += e[j]; }
    red[tid] = ls; __syncthreads();
    for (int s2 = 128; s2 > 0; s2 >>= 1) {
        if (tid < s2) red[tid] += red[tid+s2];
        __syncthreads();
    }
    const float inv = 1.0f / red[0];
    #pragma unroll
    for (int j = 0; j < 4; ++j)
        A16[(size_t)n*1024 + tid + j*256] = (_Float16)(e[j] * inv);
}

// ---------------------------------------------------------------------------
// K_w: one-shot f16 transposition of all projection weights.
//  QKVOT: [3][4][64][64]  (q,k,v,o; WT[c][k] = W[k][c])
//  W1T:   [3][256][64], W2T: [3][64][256], WcT: [3][64][256]
// ---------------------------------------------------------------------------
__global__ void k_wprep(const float* __restrict__ Wq, const float* __restrict__ Wk,
                        const float* __restrict__ Wv, const float* __restrict__ Wo,
                        const float* __restrict__ W1, const float* __restrict__ W2,
                        const float* __restrict__ Wc,
                        _Float16* __restrict__ QKVOT, _Float16* __restrict__ W1T,
                        _Float16* __restrict__ W2T, _Float16* __restrict__ WcT)
{
    const int id = blockIdx.x*256 + threadIdx.x;     // 196608 total
    const int l = id >> 16;
    const int r = id & 65535;
    const int region = r >> 14;
    const int e = r & 16383;
    if (region == 0) {
        const int mat = e >> 12, ee = e & 4095;
        const int c = ee >> 6, k = ee & 63;
        const float* src = (mat==0)?Wq:(mat==1)?Wk:(mat==2)?Wv:Wo;
        QKVOT[(((size_t)l*4 + mat)*64 + c)*64 + k] = (_Float16)src[(size_t)l*4096 + k*64 + c];
    } else if (region == 1) {
        const int j = e >> 6, k = e & 63;
        W1T[((size_t)l*256 + j)*64 + k] = (_Float16)W1[(size_t)l*16384 + k*256 + j];
    } else if (region == 2) {
        const int c = e >> 8, j = e & 255;
        W2T[((size_t)l*64 + c)*256 + j] = (_Float16)W2[(size_t)l*16384 + j*64 + c];
    } else {
        const int c = e >> 8, k = e & 255;
        WcT[((size_t)l*64 + c)*256 + k] = (_Float16)Wc[(size_t)l*16384 + k*64 + c];
    }
}

// ---------------------------------------------------------------------------
// K2: Y = alpha*(A @ X) + beta*Z via v_wmma_f32_16x16x32_f16.
// A [1024][1024] f16 row-major; X channel-major f16 XT[s][64][1024].
// Outputs (all nullable): Yf f32 [s][n][64], YT f16 ch-major, Yh f16 row-major
// (stride 256; pointer pre-offset to CCAT column base).
// ---------------------------------------------------------------------------
__global__ void k_gemmAX(const _Float16* __restrict__ A16,
                         const _Float16* __restrict__ XT,
                         const float* __restrict__ Z,
                         float alpha, float beta,
                         float* __restrict__ Yf, _Float16* __restrict__ YT,
                         _Float16* __restrict__ Yh)
{
    const int lane   = threadIdx.x & 31;
    const int waveId = blockIdx.x * 8 + (threadIdx.x >> 5);   // 0..24575
    const int s    = waveId >> 8;
    const int rem  = waveId & 255;
    const int rowT = rem >> 2, colT = rem & 3;

    const int mrow = rowT*16 + (lane & 15);
    const int cg   = colT*16 + (lane & 15);
    const int hiA  = (lane & 16) ? 8  : 0;
    const int hiB  = (lane & 16) ? 8  : 0;

    const _Float16* __restrict__ arow = A16 + (size_t)mrow * 1024;
    const _Float16* __restrict__ brow = XT + ((size_t)s*64 + cg) * 1024;

    v8f acc = {};
    for (int kb = 0; kb < 1024; kb += 32) {
        const h8 a0 = *(const h8*)(arow + kb + hiA);
        const h8 a1 = *(const h8*)(arow + kb + hiA + 16);
        const h8 b0 = *(const h8*)(brow + kb + hiB);
        const h8 b1 = *(const h8*)(brow + kb + hiB + 16);
        v16h a, b;
        #pragma unroll
        for (int i = 0; i < 8; ++i) {
            a[i] = a0[i]; a[i+8] = a1[i];
            b[i] = b0[i]; b[i+8] = b1[i];
        }
        acc = __builtin_amdgcn_wmma_f32_16x16x32_f16(
                  false, a, false, b, (short)0, acc, false, false);
    }

    const int mBase = rowT*16 + ((lane & 16) ? 8 : 0);
    #pragma unroll
    for (int r = 0; r < 8; ++r) {
        const int gm = mBase + r;
        const size_t grow = (size_t)s*1024 + gm;
        float val = alpha * acc[r];
        if (Z)  val += beta * Z[grow*64 + cg];
        if (Yf) Yf[grow*64 + cg] = val;
        if (YT) YT[((size_t)s*64 + cg)*1024 + gm] = (_Float16)val;
        if (Yh) Yh[grow*256 + cg] = (_Float16)val;
    }
}

// ---------------------------------------------------------------------------
// K3: generic row GEMM  Y = act( A[ROWS,Cin] @ B[Cin,Cout] + bias (+ Z) )
// A f16 row-major (lda), B given pre-transposed f16 BT[Cout][Cin].
// One wave per 16x16 tile; tiles = (ROWS/16) * (Cout/16).
// ---------------------------------------------------------------------------
__global__ void k_gemmRW(const _Float16* __restrict__ Arow, int lda,
                         const _Float16* __restrict__ BT,
                         const float* __restrict__ bias,
                         const float* __restrict__ Zres,
                         int Cin, int Cout, int relu,
                         float* __restrict__ Yf,
                         _Float16* __restrict__ Yh, int ldh)
{
    const int lane   = threadIdx.x & 31;
    const int waveId = blockIdx.x * 8 + (threadIdx.x >> 5);
    const int colTiles = Cout >> 4;
    const int rowT = waveId / colTiles;
    const int colT = waveId - rowT*colTiles;

    const int mrow = rowT*16 + (lane & 15);
    const int cg   = colT*16 + (lane & 15);
    const int hiA  = (lane & 16) ? 8  : 0;
    const int hiB  = (lane & 16) ? 8  : 0;

    const _Float16* __restrict__ arow = Arow + (size_t)mrow * lda;
    const _Float16* __restrict__ brow = BT + (size_t)cg * Cin;

    v8f acc = {};
    for (int kb = 0; kb < Cin; kb += 32) {
        const h8 a0 = *(const h8*)(arow + kb + hiA);
        const h8 a1 = *(const h8*)(arow + kb + hiA + 16);
        const h8 b0 = *(const h8*)(brow + kb + hiB);
        const h8 b1 = *(const h8*)(brow + kb + hiB + 16);
        v16h a, b;
        #pragma unroll
        for (int i = 0; i < 8; ++i) {
            a[i] = a0[i]; a[i+8] = a1[i];
            b[i] = b0[i]; b[i+8] = b1[i];
        }
        acc = __builtin_amdgcn_wmma_f32_16x16x32_f16(
                  false, a, false, b, (short)0, acc, false, false);
    }

    const int mBase = rowT*16 + ((lane & 16) ? 8 : 0);
    const float bb = bias ? bias[cg] : 0.f;
    #pragma unroll
    for (int r = 0; r < 8; ++r) {
        const size_t grow = (size_t)mBase + r;
        float val = acc[r] + bb;
        if (Zres) val += Zres[grow*Cout + cg];
        if (relu) val = fmaxf(val, 0.f);
        if (Yf) Yf[grow*Cout + cg] = val;
        if (Yh) Yh[grow*ldh + cg] = (_Float16)val;
    }
}

// ---------------------------------------------------------------------------
// K4: fused Q,K,V projection (A tile loaded once, 3 accumulators).
// Cin = Cout = 64.  Outputs f16 row-major [row][64].
// ---------------------------------------------------------------------------
__global__ void k_gemmQKV(const _Float16* __restrict__ Arow,
                          const _Float16* __restrict__ WT,   // [3][64][64] q,k,v
                          const float* __restrict__ bq, const float* __restrict__ bk,
                          const float* __restrict__ bv,
                          _Float16* __restrict__ Qh, _Float16* __restrict__ Kh,
                          _Float16* __restrict__ Vh)
{
    const int lane   = threadIdx.x & 31;
    const int waveId = blockIdx.x * 8 + (threadIdx.x >> 5);   // 0..24575
    const int rowT = waveId >> 2, colT = waveId & 3;

    const int mrow = rowT*16 + (lane & 15);
    const int cg   = colT*16 + (lane & 15);
    const int hiA  = (lane & 16) ? 8  : 0;
    const int hiB  = (lane & 16) ? 8  : 0;

    const _Float16* __restrict__ arow = Arow + (size_t)mrow * 64;
    const _Float16* __restrict__ bq_r = WT + (size_t)cg * 64;
    const _Float16* __restrict__ bk_r = WT + 4096 + (size_t)cg * 64;
    const _Float16* __restrict__ bv_r = WT + 8192 + (size_t)cg * 64;

    v8f aq = {}, ak = {}, av = {};
    #pragma unroll
    for (int kb = 0; kb < 64; kb += 32) {
        const h8 a0 = *(const h8*)(arow + kb + hiA);
        const h8 a1 = *(const h8*)(arow + kb + hiA + 16);
        v16h a;
        #pragma unroll
        for (int i = 0; i < 8; ++i) { a[i] = a0[i]; a[i+8] = a1[i]; }

        const h8 q0 = *(const h8*)(bq_r + kb + hiB);
        const h8 q1 = *(const h8*)(bq_r + kb + hiB + 16);
        const h8 k0 = *(const h8*)(bk_r + kb + hiB);
        const h8 k1 = *(const h8*)(bk_r + kb + hiB + 16);
        const h8 v0 = *(const h8*)(bv_r + kb + hiB);
        const h8 v1 = *(const h8*)(bv_r + kb + hiB + 16);
        v16h bq16, bk16, bv16;
        #pragma unroll
        for (int i = 0; i < 8; ++i) {
            bq16[i] = q0[i]; bq16[i+8] = q1[i];
            bk16[i] = k0[i]; bk16[i+8] = k1[i];
            bv16[i] = v0[i]; bv16[i+8] = v1[i];
        }
        aq = __builtin_amdgcn_wmma_f32_16x16x32_f16(false, a, false, bq16, (short)0, aq, false, false);
        ak = __builtin_amdgcn_wmma_f32_16x16x32_f16(false, a, false, bk16, (short)0, ak, false, false);
        av = __builtin_amdgcn_wmma_f32_16x16x32_f16(false, a, false, bv16, (short)0, av, false, false);
    }

    const int mBase = rowT*16 + ((lane & 16) ? 8 : 0);
    const float cbq = bq[cg], cbk = bk[cg], cbv = bv[cg];
    #pragma unroll
    for (int r = 0; r < 8; ++r) {
        const size_t grow = (size_t)mBase + r;
        Qh[grow*64 + cg] = (_Float16)(aq[r] + cbq);
        Kh[grow*64 + cg] = (_Float16)(ak[r] + cbk);
        Vh[grow*64 + cg] = (_Float16)(av[r] + cbv);
    }
}

// ---------------------------------------------------------------------------
// K5: attention core for one (b,n): scores (hd=16, scale .25), softmax over
// T=12 keys, o = p @ v.  Writes Oh f16 row-major [row][64].
// ---------------------------------------------------------------------------
__global__ void k_attncore(const _Float16* __restrict__ Qh, const _Float16* __restrict__ Kh,
                           const _Float16* __restrict__ Vh, _Float16* __restrict__ Oh)
{
    __shared__ float qs[768], ks[768], vs[768], att[576];
    const int tid = threadIdx.x;
    const int bn = blockIdx.x;
    const int b = bn >> 10, n = bn & 1023;

    for (int e = tid; e < 768; e += 256) {
        const int t = e >> 6, c = e & 63;
        const size_t ri = (((size_t)(b*12 + t))*1024 + n)*64 + c;
        qs[e] = (float)Qh[ri]; ks[e] = (float)Kh[ri]; vs[e] = (float)Vh[ri];
    }
    __syncthreads();

    for (int e = tid; e < 576; e += 256) {
        const int hh = e / 144, r2 = (e % 144) / 12, kk = e % 12;
        float s2 = 0.f;
        #pragma unroll
        for (int d = 0; d < 16; ++d)
            s2 += qs[r2*64 + hh*16 + d] * ks[kk*64 + hh*16 + d];
        att[e] = s2 * 0.25f;
    }
    __syncthreads();
    for (int e = tid; e < 48; e += 256) {
        float* ar = &att[e*12];
        float mx = ar[0];
        for (int kk = 1; kk < 12; ++kk) mx = fmaxf(mx, ar[kk]);
        float sm = 0.f;
        for (int kk = 0; kk < 12; ++kk) { ar[kk] = __expf(ar[kk] - mx); sm += ar[kk]; }
        const float inv = 1.f / sm;
        for (int kk = 0; kk < 12; ++kk) ar[kk] *= inv;
    }
    __syncthreads();
    for (int e = tid; e < 768; e += 256) {
        const int t = e >> 6, c = e & 63, hh = c >> 4;
        const float* ar = &att[hh*144 + t*12];
        float a = 0.f;
        #pragma unroll
        for (int kk = 0; kk < 12; ++kk) a += ar[kk] * vs[kk*64 + c];
        Oh[(((size_t)(b*12 + t))*1024 + n)*64 + c] = (_Float16)a;
    }
}

// ---------------------------------------------------------------------------
// K6: LayerNorm over 64 channels.  One wave per row (lane -> 2 channels).
// Emits f32 (Yf), f16 row-major (Yh, stride ldh), f16 ch-major (YT) - nullable.
// ---------------------------------------------------------------------------
__global__ void k_ln(const float* __restrict__ Zin,
                     const float* __restrict__ g, const float* __restrict__ be,
                     float* __restrict__ Yf,
                     _Float16* __restrict__ Yh, int ldh,
                     _Float16* __restrict__ YT)
{
    const int w = threadIdx.x >> 5, lane = threadIdx.x & 31;
    const int row = blockIdx.x * 8 + w;
    const int s = row >> 10, n = row & 1023;
    (void)w;
    const float z0 = Zin[(size_t)row*64 + lane];
    const float z1 = Zin[(size_t)row*64 + 32 + lane];
    const float m  = wred_sum(z0 + z1) * (1.f/64.f);
    const float d0 = z0 - m, d1 = z1 - m;
    const float var = wred_sum(d0*d0 + d1*d1) * (1.f/64.f);
    const float rstd = rsqrtf(var + 1e-5f);
    const float y0 = d0*rstd*g[lane]      + be[lane];
    const float y1 = d1*rstd*g[lane+32]   + be[lane+32];
    if (Yf) { Yf[(size_t)row*64 + lane] = y0; Yf[(size_t)row*64 + 32 + lane] = y1; }
    if (Yh) { Yh[(size_t)row*ldh + lane] = (_Float16)y0; Yh[(size_t)row*ldh + 32 + lane] = (_Float16)y1; }
    if (YT) {
        YT[((size_t)s*64 + lane)*1024 + n]      = (_Float16)y0;
        YT[((size_t)s*64 + 32 + lane)*1024 + n] = (_Float16)y1;
    }
}

// ---------------------------------------------------------------------------
// K7: output projection out[b][to][n] = b_out[to] + sum_{t,c} h * W_out.
// ---------------------------------------------------------------------------
__global__ void k_out(const float* __restrict__ H, const float* __restrict__ Wout,
                      const float* __restrict__ bout, float* __restrict__ out)
{
    const int id = blockIdx.x * 256 + threadIdx.x;
    const int b = id / 12288;
    const int r = id % 12288;
    const int n = r / 12, to = r % 12;
    float a = bout[to];
    for (int t = 0; t < 12; ++t) {
        const float* hr = H + (((size_t)(b*12 + t))*1024 + n)*64;
        const float* wr = Wout + (size_t)(t*64)*12 + to;
        for (int c = 0; c < 64; ++c) a += hr[c] * wr[c*12];
    }
    out[((size_t)(b*12 + to))*1024 + n] = a;
}

// ---------------------------------------------------------------------------
extern "C" void kernel_launch(void* const* d_in, const int* in_sizes, int n_in,
                              void* d_out, int out_size, void* d_ws, size_t ws_size,
                              hipStream_t stream)
{
    const float* hist  = (const float*)d_in[0];
    const float* W_in  = (const float*)d_in[5];
    const float* b_in  = (const float*)d_in[6];
    const float* mem1  = (const float*)d_in[7];
    const float* Wqm   = (const float*)d_in[9];
    const float* bqm   = (const float*)d_in[10];
    const float* E1    = (const float*)d_in[11];
    const float* E2    = (const float*)d_in[12];
    const float* chebW = (const float*)d_in[13];
    const float* chebB = (const float*)d_in[14];
    const float* Wq_a  = (const float*)d_in[15];
    const float* bq_a  = (const float*)d_in[16];
    const float* Wk_a  = (const float*)d_in[17];
    const float* bk_a  = (const float*)d_in[18];
    const float* Wv_a  = (const float*)d_in[19];
    const float* bv_a  = (const float*)d_in[20];
    const float* Wo_a  = (const float*)d_in[21];
    const float* bo_a  = (const float*)d_in[22];
    const float* ffW1  = (const float*)d_in[23];
    const float* ffb1  = (const float*)d_in[24];
    const float* ffW2  = (const float*)d_in[25];
    const float* ffb2  = (const float*)d_in[26];
    const float* ln1g  = (const float*)d_in[27];
    const float* ln1b  = (const float*)d_in[28];
    const float* ln2g  = (const float*)d_in[29];
    const float* ln2b  = (const float*)d_in[30];
    const float* Wout  = (const float*)d_in[31];
    const float* bout  = (const float*)d_in[32];
    float* out = (float*)d_out;

    char* ws = (char*)d_ws;
    size_t off = 0;
    auto alloc = [&](size_t bytes) -> void* {
        void* p = ws + off;
        off += (bytes + 255) & ~(size_t)255;
        return p;
    };
    const size_t ROWS = (size_t)96*1024;
    const size_t F32S = ROWS*64*4;        // 25.2 MB
    const size_t F16S = ROWS*64*2;        // 12.6 MB
    _Float16* A16  = (_Float16*)alloc((size_t)1024*1024*2);
    float*    H    = (float*)   alloc(F32S);        // x0 f32 (residual source)
    _Float16* HT   = (_Float16*)alloc(F16S);        // x0 ch-major f16
    _Float16* CCAT = (_Float16*)alloc(ROWS*256*2);  // [x0|x1|x2|x3] row-major f16
    float*    X1   = (float*)   alloc(F32S);        // x1 f32 (Z for stage 3)
    _Float16* X1T  = (_Float16*)alloc(F16S);
    _Float16* X2T  = (_Float16*)alloc(F16S);
    float*    HG   = (float*)   alloc(F32S);        // cheb output f32 (residual)
    _Float16* HGh  = (_Float16*)alloc(F16S);        // cheb output f16 row-major
    _Float16* Qh   = (_Float16*)alloc(F16S);
    _Float16* Kh   = (_Float16*)alloc(F16S);
    _Float16* Vh   = (_Float16*)alloc(F16S);
    _Float16* Oh   = (_Float16*)alloc(F16S);
    float*    ZR   = (float*)   alloc(F32S);        // pre-LN residual sums
    float*    Y1   = (float*)   alloc(F32S);        // post-LN1 f32
    _Float16* Y1h  = (_Float16*)alloc(F16S);        // post-LN1 f16
    _Float16* QKVOT= (_Float16*)alloc((size_t)3*4*64*64*2);
    _Float16* W1T  = (_Float16*)alloc((size_t)3*256*64*2);
    _Float16* W2T  = (_Float16*)alloc((size_t)3*64*256*2);
    _Float16* WcT  = (_Float16*)alloc((size_t)3*64*256*2);
    // F1 (FF hidden, f16 [ROWS][256]) aliases CCAT: CCAT is consumed by the
    // cheb GEMM before FF1 runs, and rewritten next layer after FF2 reads it.
    _Float16* F1h  = CCAT;

    k_wprep<<<768, 256, 0, stream>>>(Wq_a, Wk_a, Wv_a, Wo_a, ffW1, ffW2, chebW,
                                     QKVOT, W1T, W2T, WcT);
    k_adj  <<<1024, 256, 0, stream>>>(E1, E2, A16);
    k_prep <<<12288, 256, 0, stream>>>(hist, W_in, b_in, mem1, Wqm, bqm, H, HT, CCAT);

    for (int l = 0; l < 3; ++l) {
        // Chebyshev recurrence on the matrix path (WMMA):
        k_gemmAX<<<3072, 256, 0, stream>>>(A16, HT,  nullptr, 1.f,  0.f, X1, X1T, CCAT + 64);
        k_gemmAX<<<3072, 256, 0, stream>>>(A16, X1T, H,       2.f, -1.f, nullptr, X2T, CCAT + 128);
        k_gemmAX<<<3072, 256, 0, stream>>>(A16, X2T, X1,      2.f, -1.f, nullptr, nullptr, CCAT + 192);
        // hg = relu([x0|x1|x2|x3] @ Wc + bc)       [ROWS,256]@[256,64] WMMA
        k_gemmRW<<<3072, 256, 0, stream>>>(CCAT, 256, WcT + (size_t)l*64*256,
                                           chebB + (size_t)l*64, nullptr,
                                           256, 64, 1, HG, HGh, 64);
        // fused Q,K,V                              [ROWS,64]@[64,64]x3 WMMA
        k_gemmQKV<<<3072, 256, 0, stream>>>(HGh, QKVOT + (size_t)l*4*4096,
                                            bq_a + (size_t)l*64, bk_a + (size_t)l*64,
                                            bv_a + (size_t)l*64, Qh, Kh, Vh);
        // scores/softmax/o (T=12, tiny)
        k_attncore<<<8192, 256, 0, stream>>>(Qh, Kh, Vh, Oh);
        // o-proj + residual                        [ROWS,64]@[64,64] WMMA
        k_gemmRW<<<3072, 256, 0, stream>>>(Oh, 64, QKVOT + (size_t)l*4*4096 + 3*4096,
                                           bo_a + (size_t)l*64, HG,
                                           64, 64, 0, ZR, nullptr, 0);
        k_ln<<<12288, 256, 0, stream>>>(ZR, ln1g + (size_t)l*64, ln1b + (size_t)l*64,
                                        Y1, Y1h, 64, nullptr);
        // FF1 (relu)                               [ROWS,64]@[64,256] WMMA
        k_gemmRW<<<12288, 256, 0, stream>>>(Y1h, 64, W1T + (size_t)l*256*64,
                                            ffb1 + (size_t)l*256, nullptr,
                                            64, 256, 1, nullptr, F1h, 256);
        // FF2 + residual                           [ROWS,256]@[256,64] WMMA
        k_gemmRW<<<3072, 256, 0, stream>>>(F1h, 256, W2T + (size_t)l*64*256,
                                           ffb2 + (size_t)l*64, Y1,
                                           256, 64, 0, ZR, nullptr, 0);
        // LN2 -> new h: f32 (H), CCAT col0 (row-major f16), HT (ch-major f16)
        k_ln<<<12288, 256, 0, stream>>>(ZR, ln2g + (size_t)l*64, ln2b + (size_t)l*64,
                                        H, CCAT, 256, HT);
    }
    k_out<<<384, 256, 0, stream>>>(H, Wout, bout, out);

    (void)in_sizes; (void)n_in; (void)out_size; (void)ws_size;
}
